// IORTransformer_67130338836758
// MI455X (gfx1250) — matrix operand order repaired and checked by reading, not compiled, not run on hardware
//
#include <hip/hip_runtime.h>
#include <hip/hip_bf16.h>
#include <math.h>

typedef __attribute__((ext_vector_type(16))) __bf16 v16bf;
typedef __attribute__((ext_vector_type(8)))  float  v8f;

#define DIM   512
#define LSEQ  4096
#define NB    32
#define NH    8
#define HD    64
#define FFN   2048
#define TM    64          // rows of mem per workgroup
#define SA_STRIDE 520     // ushort stride for bf16 activation tile (conflict-free)
#define SC_STRIDE 516     // float stride for f32 staging tile
#define LN_EPS 1e-5f

// ---- helpers -------------------------------------------------------------

__device__ __forceinline__ unsigned short f2bf(float f) {
  unsigned int u = __float_as_uint(f);
  u += 0x7fffu + ((u >> 16) & 1u);     // round-to-nearest-even
  return (unsigned short)(u >> 16);
}

// 2x2 register-blocked GEMM: two 16-row tiles x two 16-col tiles, K=512 via
// 16 steps of v_wmma_f32_16x16x32_bf16. Each K step: 2 A frags (LDS) +
// 2 B frags (global, pre-swizzled) feed 4 WMMAs -> 2x operand reuse.
__device__ __forceinline__ void gemm_block_2x2(
    const unsigned short* __restrict__ smA,
    const unsigned short* __restrict__ Bswz,
    int NT, int mt0, int nt0, int lane, v8f acc[4])   // acc[mi*2+ni]
{
  const int g  = lane >> 4;
  const int nl = lane & 15;
#pragma unroll
  for (int kt = 0; kt < 16; ++kt) {
    union { v16bf v; unsigned int u[8]; } a0, a1;
    union { v16bf v; uint4 q[2]; } b0, b1;
    // A frags: lane holds row M=nl of tile, K-group g (e<8 -> K=g*8+e,
    // e>=8 -> K=16+g*8+(e-8)); consecutive K pairs => dword LDS loads.
    const unsigned short* ap0 = smA + (mt0 * 16 + nl) * SA_STRIDE + kt * 32 + g * 8;
    const unsigned short* ap1 = ap0 + 16 * SA_STRIDE;
#pragma unroll
    for (int i = 0; i < 4; ++i) {
      a0.u[i]     = *(const unsigned int*)(ap0 + 2 * i);
      a0.u[4 + i] = *(const unsigned int*)(ap0 + 16 + 2 * i);
      a1.u[i]     = *(const unsigned int*)(ap1 + 2 * i);
      a1.u[4 + i] = *(const unsigned int*)(ap1 + 16 + 2 * i);
    }
    // B frags: 32 contiguous bytes per lane, adjacent n-tiles.
    const uint4* bp0 = (const uint4*)(Bswz + ((size_t)(kt * NT + nt0) * 32 + lane) * 16);
    const uint4* bp1 = (const uint4*)(Bswz + ((size_t)(kt * NT + nt0 + 1) * 32 + lane) * 16);
    b0.q[0] = bp0[0]; b0.q[1] = bp0[1];
    b1.q[0] = bp1[0]; b1.q[1] = bp1[1];
    acc[0] = __builtin_amdgcn_wmma_f32_16x16x32_bf16(false, a0.v, false, b0.v, (short)0, acc[0], false, false);
    acc[1] = __builtin_amdgcn_wmma_f32_16x16x32_bf16(false, a0.v, false, b1.v, (short)0, acc[1], false, false);
    acc[2] = __builtin_amdgcn_wmma_f32_16x16x32_bf16(false, a1.v, false, b0.v, (short)0, acc[2], false, false);
    acc[3] = __builtin_amdgcn_wmma_f32_16x16x32_bf16(false, a1.v, false, b1.v, (short)0, acc[3], false, false);
  }
}

// ---- prep kernels --------------------------------------------------------

__global__ void k_zero(float* p, int n) {
  int i = blockIdx.x * blockDim.x + threadIdx.x;
  if (i < n) p[i] = 0.f;
}

// Swizzle f32 (K,N) row-major weights into per-lane bf16 WMMA B fragments.
// B layout (16-bit, 32x16): lane = N + 16*g; element e<8 -> K = g*8+e, e>=8 -> K = 16+g*8+(e-8).
__global__ void k_swz(const float* __restrict__ W, unsigned short* __restrict__ dst,
                      int N, int nTiles) {
  int id = blockIdx.x * 256 + threadIdx.x;
  if (id >= nTiles * 32) return;
  int t = id >> 5, lane = id & 31;
  int NT = N >> 4;
  int kt = t / NT, nt = t % NT;
  int g = lane >> 4, n = lane & 15;
  const float* src = W + (size_t)(kt * 32 + g * 8) * N + nt * 16 + n;
  unsigned short* d = dst + ((size_t)t * 32 + lane) * 16;
#pragma unroll
  for (int e = 0; e < 16; ++e) d[e] = f2bf(src[(size_t)(e + (e >> 3) * 8) * N]);
}

// q = LN(query) @ Wq + bq   (per batch row)
__global__ void k_query(const float* __restrict__ query, const float* __restrict__ g_q,
                        const float* __restrict__ be_q, const float* __restrict__ Wq,
                        const float* __restrict__ bq, float* __restrict__ qbuf) {
  __shared__ float sq[DIM];
  __shared__ float sred[512];
  int b = blockIdx.x, t = threadIdx.x;
  float v1 = query[b * DIM + t];
  float v2 = query[b * DIM + 256 + t];
  sred[t] = v1 + v2;
  sred[256 + t] = v1 * v1 + v2 * v2;
  __syncthreads();
  for (int s = 128; s > 0; s >>= 1) {
    if (t < s) { sred[t] += sred[t + s]; sred[256 + t] += sred[256 + t + s]; }
    __syncthreads();
  }
  float mu  = sred[0] * (1.f / DIM);
  float var = sred[256] * (1.f / DIM) - mu * mu;
  float inv = rsqrtf(var + LN_EPS);
  sq[t]       = (v1 - mu) * inv * g_q[t]       + be_q[t];
  sq[256 + t] = (v2 - mu) * inv * g_q[256 + t] + be_q[256 + t];
  __syncthreads();
  for (int half = 0; half < 2; ++half) {
    int c = t + half * 256;
    float acc = bq[c];
    for (int k = 0; k < DIM; ++k) acc += sq[k] * Wq[(size_t)k * DIM + c];
    qbuf[b * DIM + c] = acc;
  }
}

// ---- fused main kernel ---------------------------------------------------
// Per WG (64 rows of one batch): mem->bf16 LDS, GEMM1(WMMA)+relu+LN ->bf16,
// GEMM2 k-half, scores, GEMM2 v-half, weighted sum -> atomic partial x.
__global__ __launch_bounds__(256, 1)
void k_main(const float* __restrict__ mem, const float* __restrict__ ior,
            const float* __restrict__ b_in, const float* __restrict__ g_in,
            const float* __restrict__ be_in,
            const unsigned short* __restrict__ winS,
            const unsigned short* __restrict__ wkvS,
            const float* __restrict__ bkv,
            const float* __restrict__ qbuf,
            float* __restrict__ xacc) {
  extern __shared__ char smraw[];
  unsigned short* smA = (unsigned short*)smraw;                               // 64*520 bf16
  float* smC    = (float*)(smraw + TM * SA_STRIDE * 2);                       // 64*516 f32
  float* smS    = (float*)(smraw + TM * SA_STRIDE * 2 + TM * SC_STRIDE * 4);  // 512
  float* smQ    = smS + 512;                                                  // 512
  float* smRed  = smQ + 512;                                                  // 512
  float* smStats = smRed + 512;                                               // 128

  const int tid = threadIdx.x;
  const int lane = tid & 31, wave = tid >> 5;
  const int g = lane >> 4, nl = lane & 15;
  const int gRow0 = blockIdx.x * TM;
  const int b  = gRow0 >> 12;            // 4096 rows per batch
  const int l0 = gRow0 & (LSEQ - 1);

  // Phase 0: mem tile -> bf16 LDS; q row -> LDS
  const float4* memB = (const float4*)(mem + (size_t)gRow0 * DIM);
#pragma unroll
  for (int i = 0; i < 32; ++i) {
    int e4 = i * 256 + tid;
    int row = e4 >> 7;
    int c4 = (e4 & 127) << 2;
    float4 f = memB[e4];
    unsigned int lo = (unsigned)f2bf(f.x) | ((unsigned)f2bf(f.y) << 16);
    unsigned int hi = (unsigned)f2bf(f.z) | ((unsigned)f2bf(f.w) << 16);
    *(uint2*)(smA + row * SA_STRIDE + c4) = make_uint2(lo, hi);
  }
  smQ[tid]       = qbuf[b * DIM + tid];
  smQ[tid + 256] = qbuf[b * DIM + 256 + tid];
  __syncthreads();

  // Phase 1: C1 = relu(mem @ W_in + b_in)  (f32 in smC)
  // 32 blocks of 2x2 tiles (mt0 in {0,2}, nt0 even in 0..30) over 8 waves.
  for (int q = 0; q < 4; ++q) {
    int bb = wave * 4 + q;
    int mt0 = (bb & 1) * 2;
    int nt0 = (bb >> 1) * 2;
    v8f acc[4];
#pragma unroll
    for (int i = 0; i < 4; ++i) acc[i] = (v8f){0.f,0.f,0.f,0.f,0.f,0.f,0.f,0.f};
    gemm_block_2x2(smA, winS, 32, mt0, nt0, lane, acc);
#pragma unroll
    for (int mi = 0; mi < 2; ++mi) {
#pragma unroll
      for (int ni = 0; ni < 2; ++ni) {
        int col = (nt0 + ni) * 16 + nl;
        float bi = b_in[col];
        v8f a = acc[mi * 2 + ni];
#pragma unroll
        for (int r = 0; r < 8; ++r) {
          int row = (mt0 + mi) * 16 + g * 8 + r;   // C/D: M = r + 8*(lane>=16)
          smC[row * SC_STRIDE + col] = fmaxf(a[r] + bi, 0.f);
        }
      }
    }
  }
  __syncthreads();

  // Phase 2: per-row LN stats
  {
    int part = tid & 3, row = tid >> 2;
    const float* rp = smC + row * SC_STRIDE + part * 128;
    float s = 0.f, s2 = 0.f;
    for (int c = 0; c < 128; ++c) { float x = rp[c]; s += x; s2 += x * x; }
    smRed[row * 4 + part] = s;
    smRed[256 + row * 4 + part] = s2;
  }
  __syncthreads();
  if (tid < 64) {
    float s  = smRed[tid * 4] + smRed[tid * 4 + 1] + smRed[tid * 4 + 2] + smRed[tid * 4 + 3];
    float s2 = smRed[256 + tid * 4] + smRed[257 + tid * 4] + smRed[258 + tid * 4] + smRed[259 + tid * 4];
    float mu  = s * (1.f / DIM);
    float var = s2 * (1.f / DIM) - mu * mu;
    smStats[tid * 2]     = mu;
    smStats[tid * 2 + 1] = rsqrtf(var + LN_EPS);
  }
  __syncthreads();

  // Phase 3: normalize, scale/shift, requantize bf16 into smA (GEMM2 A)
  for (int i = 0; i < 32; ++i) {
    int e4 = i * 256 + tid;
    int row = e4 >> 7;
    int c4 = (e4 & 127) << 2;
    float mu = smStats[row * 2], inv = smStats[row * 2 + 1];
    float o[4];
#pragma unroll
    for (int j = 0; j < 4; ++j) {
      int c = c4 + j;
      o[j] = (smC[row * SC_STRIDE + c] - mu) * inv * g_in[c] + be_in[c];
    }
    unsigned int lo = (unsigned)f2bf(o[0]) | ((unsigned)f2bf(o[1]) << 16);
    unsigned int hi = (unsigned)f2bf(o[2]) | ((unsigned)f2bf(o[3]) << 16);
    *(uint2*)(smA + row * SA_STRIDE + c4) = make_uint2(lo, hi);
  }
  __syncthreads();

  // Phase 4: k = m @ Wkv[:, :512] + bkv[:512]   (f32 into smC)
  for (int q = 0; q < 4; ++q) {
    int bb = wave * 4 + q;
    int mt0 = (bb & 1) * 2;
    int nt0 = (bb >> 1) * 2;                      // nt in 0..31 (k half)
    v8f acc[4];
#pragma unroll
    for (int i = 0; i < 4; ++i) acc[i] = (v8f){0.f,0.f,0.f,0.f,0.f,0.f,0.f,0.f};
    gemm_block_2x2(smA, wkvS, 64, mt0, nt0, lane, acc);
#pragma unroll
    for (int mi = 0; mi < 2; ++mi) {
#pragma unroll
      for (int ni = 0; ni < 2; ++ni) {
        int col = (nt0 + ni) * 16 + nl;
        float bi = bkv[col];
        v8f a = acc[mi * 2 + ni];
#pragma unroll
        for (int r = 0; r < 8; ++r) {
          int row = (mt0 + mi) * 16 + g * 8 + r;
          smC[row * SC_STRIDE + col] = a[r] + bi;
        }
      }
    }
  }
  __syncthreads();

  // Phase 5: s[l,hd] = scale * ior[b,hd,l0+l] * dot(q[hd], k[l,hd])
  for (int h2 = 0; h2 < 2; ++h2) {
    int p = tid + h2 * 256;
    int l = p >> 3, hd = p & 7;
    const float* kp = smC + l * SC_STRIDE + hd * HD;
    const float* qp = smQ + hd * HD;
    float s = 0.f;
#pragma unroll 8
    for (int d = 0; d < HD; ++d) s += kp[d] * qp[d];
    s *= 0.125f * ior[((size_t)(b * NH + hd)) * LSEQ + l0 + l];
    smS[l * 8 + hd] = s;
  }
  __syncthreads();

  // Phase 6: v = m @ Wkv[:, 512:] + bkv[512:]  (f32 into smC, shifted)
  for (int q = 0; q < 4; ++q) {
    int bb = wave * 4 + q;
    int mt0 = (bb & 1) * 2;
    int nt0 = 32 + (bb >> 1) * 2;                 // nt in 32..63 (v half)
    v8f acc[4];
#pragma unroll
    for (int i = 0; i < 4; ++i) acc[i] = (v8f){0.f,0.f,0.f,0.f,0.f,0.f,0.f,0.f};
    gemm_block_2x2(smA, wkvS, 64, mt0, nt0, lane, acc);
#pragma unroll
    for (int mi = 0; mi < 2; ++mi) {
#pragma unroll
      for (int ni = 0; ni < 2; ++ni) {
        int col = (nt0 + ni) * 16 + nl;           // 512..1023
        float bi = bkv[col];
        v8f a = acc[mi * 2 + ni];
#pragma unroll
        for (int r = 0; r < 8; ++r) {
          int row = (mt0 + mi) * 16 + g * 8 + r;
          smC[row * SC_STRIDE + (col - 512)] = a[r] + bi;
        }
      }
    }
  }
  __syncthreads();

  // Phase 7: xpart[c] = sum_l s[l, c/64] * v[l, c]; atomic accumulate
  for (int h2 = 0; h2 < 2; ++h2) {
    int c = tid + h2 * 256;
    int hd = c >> 6;
    float acc = 0.f;
    for (int l = 0; l < TM; ++l) acc += smS[l * 8 + hd] * smC[l * SC_STRIDE + c];
    atomicAdd(&xacc[b * DIM + c], acc);
  }
}

// ---- epilogue: x += query; FFN with exact GELU; residual -----------------
__global__ void k_epi(const float* __restrict__ xacc, const float* __restrict__ query,
                      const float* __restrict__ g_f, const float* __restrict__ be_f,
                      const float* __restrict__ W1, const float* __restrict__ b1,
                      const float* __restrict__ W2, const float* __restrict__ b2,
                      float* __restrict__ out) {
  __shared__ float sx[DIM], st[DIM], shid[FFN], sred[512];
  int b = blockIdx.x, t = threadIdx.x;
  float x1 = xacc[b * DIM + t]       + query[b * DIM + t];
  float x2 = xacc[b * DIM + 256 + t] + query[b * DIM + 256 + t];
  sx[t] = x1; sx[256 + t] = x2;
  sred[t] = x1 + x2;
  sred[256 + t] = x1 * x1 + x2 * x2;
  __syncthreads();
  for (int s = 128; s > 0; s >>= 1) {
    if (t < s) { sred[t] += sred[t + s]; sred[256 + t] += sred[256 + t + s]; }
    __syncthreads();
  }
  float mu  = sred[0] * (1.f / DIM);
  float var = sred[256] * (1.f / DIM) - mu * mu;
  float inv = rsqrtf(var + LN_EPS);
  st[t]       = (x1 - mu) * inv * g_f[t]       + be_f[t];
  st[256 + t] = (x2 - mu) * inv * g_f[256 + t] + be_f[256 + t];
  __syncthreads();
  for (int j = 0; j < 8; ++j) {
    int u = t + j * 256;
    float acc = b1[u];
    for (int k = 0; k < DIM; ++k) acc += st[k] * W1[(size_t)k * FFN + u];
    shid[u] = 0.5f * acc * (1.f + erff(acc * 0.70710678118654752f));
  }
  __syncthreads();
  for (int h2 = 0; h2 < 2; ++h2) {
    int c = t + h2 * 256;
    float acc = b2[c];
    for (int k = 0; k < FFN; ++k) acc += shid[k] * W2[(size_t)k * DIM + c];
    out[b * DIM + c] = sx[c] + acc;
  }
}

// ---- launch --------------------------------------------------------------

extern "C" void kernel_launch(void* const* d_in, const int* in_sizes, int n_in,
                              void* d_out, int out_size, void* d_ws, size_t ws_size,
                              hipStream_t stream) {
  (void)in_sizes; (void)n_in; (void)out_size; (void)ws_size;
  const float* query = (const float*)d_in[0];
  const float* mem   = (const float*)d_in[1];
  const float* ior   = (const float*)d_in[2];
  const float* W_in  = (const float*)d_in[3];
  const float* b_in  = (const float*)d_in[4];
  const float* g_in  = (const float*)d_in[5];
  const float* be_in = (const float*)d_in[6];
  const float* Wq    = (const float*)d_in[7];
  const float* bq    = (const float*)d_in[8];
  const float* Wkv   = (const float*)d_in[9];
  const float* bkv   = (const float*)d_in[10];
  const float* W1    = (const float*)d_in[11];
  const float* b1    = (const float*)d_in[12];
  const float* W2    = (const float*)d_in[13];
  const float* b2    = (const float*)d_in[14];
  const float* g_q   = (const float*)d_in[15];
  const float* be_q  = (const float*)d_in[16];
  const float* g_f   = (const float*)d_in[17];
  const float* be_f  = (const float*)d_in[18];
  float* out = (float*)d_out;

  char* ws = (char*)d_ws;
  float* xacc = (float*)ws;                               // 32*512 f32
  float* qbuf = (float*)(ws + 65536);                     // 32*512 f32
  unsigned short* winS = (unsigned short*)(ws + 131072);  // 512*512 bf16
  unsigned short* wkvS = (unsigned short*)(ws + 655360);  // 512*1024 bf16

  k_zero<<<64, 256, 0, stream>>>(xacc, NB * DIM);
  k_swz<<<64, 256, 0, stream>>>(W_in, winS, DIM, 16 * 32);
  k_swz<<<128, 256, 0, stream>>>(Wkv, wkvS, 2 * DIM, 16 * 64);
  k_query<<<NB, 256, 0, stream>>>(query, g_q, be_q, Wq, bq, qbuf);

  size_t shmem = (size_t)TM * SA_STRIDE * 2 + (size_t)TM * SC_STRIDE * 4 +
                 (512 + 512 + 512 + 128) * sizeof(float);   // ~200 KB (WGP has 320 KB)
  k_main<<<(NB * LSEQ) / TM, 256, shmem, stream>>>(mem, ior, b_in, g_in, be_in,
                                                   winS, wkvS, bkv, qbuf, xacc);
  k_epi<<<NB, 256, 0, stream>>>(xacc, query, g_f, be_f, W1, b1, W2, b2, out);
}
